// FeatureEmbedding_32676111188571
// MI455X (gfx1250) — matrix operand order repaired and checked by reading, not compiled, not run on hardware
//
#include <hip/hip_runtime.h>

// ---------------------------------------------------------------------------
// out[2048,512] = (x[2048,20000] @ E[20000,512]) * (1/20000) + bias
//
// Pass 1: split fp32 -> bf16 hi/lo once into workspace (x row-major,
//         E transposed to [n][k]) so the GEMM loop does zero conversions.
// Pass 2: bf16x3 WMMA GEMM, split-K=2 (gridDim.z), tiles staged global->LDS
//         with global_load_async_to_lds_b128 (ASYNCcnt), double-buffered.
// Pass 3: partial reduction + scale + bias.
// Fallback (small ws): fused convert-in-loop kernel.
// ---------------------------------------------------------------------------

typedef __attribute__((ext_vector_type(16))) __bf16 v16bf;
typedef __attribute__((ext_vector_type(8)))  __bf16 bf16x8;
typedef __attribute__((ext_vector_type(8)))  float  v8f;

constexpr int M_DIM = 2048;
constexpr int N_DIM = 512;
constexpr int K_DIM = 20000;

constexpr int BLK_M   = 128;
constexpr int BLK_N   = 64;
constexpr int BLK_K   = 32;
constexpr int THREADS = 256;               // 8 waves (wave32)
constexpr int K_ITERS = K_DIM / BLK_K;     // 625 (exact)
constexpr int LDS_K   = 40;                // padded stride: 80B rows, 16B aligned
constexpr int SPLIT_K = 2;                 // gridDim.z halves of the K chain

constexpr size_t A_ELEMS   = (size_t)M_DIM * K_DIM;   // 40,960,000
constexpr size_t B_ELEMS   = (size_t)N_DIM * K_DIM;   // 10,240,000
constexpr size_t OUT_ELEMS = (size_t)M_DIM * N_DIM;   // 1,048,576
constexpr size_t WS_BF16   = (2 * A_ELEMS + 2 * B_ELEMS) * sizeof(__bf16);
constexpr size_t WS_NEED   = WS_BF16 + SPLIT_K * OUT_ELEMS * sizeof(float);

union FragBF {
  v16bf  v;
  bf16x8 h[2];
};

struct alignas(16) SMem {
  __bf16 Ahi[2][BLK_M * LDS_K];
  __bf16 Alo[2][BLK_M * LDS_K];
  __bf16 Bhi[2][BLK_N * LDS_K];   // transposed image: [n][k]
  __bf16 Blo[2][BLK_N * LDS_K];
};  // 61440 bytes

// ---------------------------------------------------------------------------
// gfx1250 async copy helpers
// ---------------------------------------------------------------------------
__device__ __forceinline__ void async_b128(uint32_t lds_byte_addr,
                                           const void* gptr) {
  // GV mode: global address from a 64-bit VGPR pair; LDS byte address from the VDST VGPR
  asm volatile("global_load_async_to_lds_b128 %0, %1, off"
               :: "v"(lds_byte_addr), "v"(gptr)
               : "memory");
}

__device__ __forceinline__ void wait_async0() {
  asm volatile("s_wait_asynccnt 0" ::: "memory");
}

// ---------------------------------------------------------------------------
// Pass 1a: elementwise split of x into bf16 hi/lo (8 elems / thread)
// ---------------------------------------------------------------------------
__global__ __launch_bounds__(256)
void split_x_kernel(const float* __restrict__ src,
                    __bf16* __restrict__ hi, __bf16* __restrict__ lo) {
  const size_t i0 = ((size_t)blockIdx.x * 256 + threadIdx.x) * 8;
  const float4 u = ((const float4*)(src + i0))[0];
  const float4 v = ((const float4*)(src + i0))[1];
  const float f[8] = {u.x, u.y, u.z, u.w, v.x, v.y, v.z, v.w};
  bf16x8 hv, lv;
#pragma unroll
  for (int j = 0; j < 8; ++j) {
    const __bf16 h = (__bf16)f[j];
    hv[j] = h;
    lv[j] = (__bf16)(f[j] - (float)h);
  }
  *(bf16x8*)(hi + i0) = hv;
  *(bf16x8*)(lo + i0) = lv;
}

// ---------------------------------------------------------------------------
// Pass 1b: transpose E [20000,512] -> Et [512,20000] with bf16 hi/lo split.
// ---------------------------------------------------------------------------
__global__ __launch_bounds__(256)
void split_et_kernel(const float* __restrict__ E,
                     __bf16* __restrict__ hi, __bf16* __restrict__ lo) {
  __shared__ float tile[32][65];
  const int k0 = blockIdx.x * 32;
  const int n0 = blockIdx.y * 64;
  const int t  = threadIdx.x;

  {
    const int nl = t & 63;
    const int kg = (t >> 6) * 8;
#pragma unroll
    for (int j = 0; j < 8; ++j)
      tile[kg + j][nl] = E[(size_t)(k0 + kg + j) * N_DIM + n0 + nl];
  }
  __syncthreads();
  {
    const int n  = t >> 2;
    const int kc = (t & 3) * 8;
    bf16x8 hv, lv;
#pragma unroll
    for (int j = 0; j < 8; ++j) {
      const float f = tile[kc + j][n];
      const __bf16 h = (__bf16)f;
      hv[j] = h;
      lv[j] = (__bf16)(f - (float)h);
    }
    const size_t o = (size_t)(n0 + n) * K_DIM + k0 + kc;
    *(bf16x8*)(hi + o) = hv;
    *(bf16x8*)(lo + o) = lv;
  }
}

// ---------------------------------------------------------------------------
// Pass 2: bf16x3 WMMA GEMM, async staging, double-buffered, split-K.
// Writes raw fp32 partial sums: partials[z][m][n].
// ---------------------------------------------------------------------------
__global__ __launch_bounds__(THREADS)
void gemm_bf16_async_kernel(const __bf16* __restrict__ Ahi,
                            const __bf16* __restrict__ Alo,
                            const __bf16* __restrict__ Bhi,
                            const __bf16* __restrict__ Blo,
                            float* __restrict__ partials) {
  __shared__ SMem sm;

  const int t    = threadIdx.x;
  const int lane = t & 31;
  const int wid  = t >> 5;
  const int wm   = wid >> 1;      // 0..3 -> 32-row band
  const int wn   = wid & 1;       // 0..1 -> 32-col band
  const int mIn  = lane & 15;
  const int half = lane >> 4;

  const int bx = blockIdx.x;      // N tile (8)
  const int by = blockIdx.y;      // M tile (16)
  const int bz = blockIdx.z;      // K split (2)
  const int mBase = by * BLK_M;
  const int nBase = bx * BLK_N;
  const int it0 = (bz == 0) ? 0 : 313;      // 625 = 313 + 312
  const int cnt = (bz == 0) ? 313 : 312;

  uint32_t ldsAhi[2], ldsAlo[2], ldsBhi[2], ldsBlo[2];
#pragma unroll
  for (int b = 0; b < 2; ++b) {
    ldsAhi[b] = (uint32_t)(uintptr_t)&sm.Ahi[b][0];
    ldsAlo[b] = (uint32_t)(uintptr_t)&sm.Alo[b][0];
    ldsBhi[b] = (uint32_t)(uintptr_t)&sm.Bhi[b][0];
    ldsBlo[b] = (uint32_t)(uintptr_t)&sm.Blo[b][0];
  }

  v8f acc[2][2];
  acc[0][0] = {}; acc[0][1] = {}; acc[1][0] = {}; acc[1][1] = {};

  // A: 128x32 bf16 = 512 16B chunks (2/thread). B: 64x32 = 256 chunks (1/thread).
  auto issue_async = [&](int buf, int it) {
    const size_t k0 = (size_t)it * BLK_K;
#pragma unroll
    for (int c = 0; c < 2; ++c) {
      const int ch  = t + c * THREADS;
      const int row = ch >> 2;
      const int off = (ch & 3) * 8;               // bf16 elems (8 per 16B)
      const size_t g = (size_t)(mBase + row) * K_DIM + k0 + off;
      const uint32_t l = (uint32_t)(row * LDS_K + off) * 2;
      async_b128(ldsAhi[buf] + l, Ahi + g);
      async_b128(ldsAlo[buf] + l, Alo + g);
    }
    {
      const int row = t >> 2;
      const int off = (t & 3) * 8;
      const size_t g = (size_t)(nBase + row) * K_DIM + k0 + off;
      const uint32_t l = (uint32_t)(row * LDS_K + off) * 2;
      async_b128(ldsBhi[buf] + l, Bhi + g);
      async_b128(ldsBlo[buf] + l, Blo + g);
    }
  };

  auto compute = [&](int buf) {
    const __bf16* pAhi = sm.Ahi[buf];
    const __bf16* pAlo = sm.Alo[buf];
    const __bf16* pBhi = sm.Bhi[buf];
    const __bf16* pBlo = sm.Blo[buf];

    FragBF ah[2], al[2], bh[2], bl[2];
#pragma unroll
    for (int mi = 0; mi < 2; ++mi) {
      const int r = wm * 32 + mi * 16 + mIn;
      const __bf16* ph = pAhi + r * LDS_K + half * 8;
      ah[mi].h[0] = *(const bf16x8*)(ph);
      ah[mi].h[1] = *(const bf16x8*)(ph + 16);
      const __bf16* pl = pAlo + r * LDS_K + half * 8;
      al[mi].h[0] = *(const bf16x8*)(pl);
      al[mi].h[1] = *(const bf16x8*)(pl + 16);
    }
#pragma unroll
    for (int ni = 0; ni < 2; ++ni) {
      const int n = wn * 32 + ni * 16 + mIn;
      const __bf16* ph = pBhi + n * LDS_K + half * 8;
      bh[ni].h[0] = *(const bf16x8*)(ph);
      bh[ni].h[1] = *(const bf16x8*)(ph + 16);
      const __bf16* pl = pBlo + n * LDS_K + half * 8;
      bl[ni].h[0] = *(const bf16x8*)(pl);
      bl[ni].h[1] = *(const bf16x8*)(pl + 16);
    }
#pragma unroll
    for (int mi = 0; mi < 2; ++mi) {
#pragma unroll
      for (int ni = 0; ni < 2; ++ni) {
        acc[mi][ni] = __builtin_amdgcn_wmma_f32_16x16x32_bf16(
            false, ah[mi].v, false, bh[ni].v, (short)0, acc[mi][ni], false, false);
        acc[mi][ni] = __builtin_amdgcn_wmma_f32_16x16x32_bf16(
            false, ah[mi].v, false, bl[ni].v, (short)0, acc[mi][ni], false, false);
        acc[mi][ni] = __builtin_amdgcn_wmma_f32_16x16x32_bf16(
            false, al[mi].v, false, bh[ni].v, (short)0, acc[mi][ni], false, false);
      }
    }
  };

  issue_async(0, it0);
  wait_async0();
  __syncthreads();

  for (int i = 0; i < cnt; ++i) {
    const int nxt = i + 1;
    if (nxt < cnt) issue_async(nxt & 1, it0 + nxt);   // in flight under WMMAs
    compute(i & 1);
    wait_async0();
    __syncthreads();
  }

  float* P = partials + (size_t)bz * OUT_ELEMS;
#pragma unroll
  for (int mi = 0; mi < 2; ++mi) {
#pragma unroll
    for (int ni = 0; ni < 2; ++ni) {
      const int row0 = mBase + wm * 32 + mi * 16 + half * 8;
      const int col  = nBase + wn * 32 + ni * 16 + mIn;
#pragma unroll
      for (int j = 0; j < 8; ++j)
        P[(size_t)(row0 + j) * N_DIM + col] = acc[mi][ni][j];
    }
  }
}

// ---------------------------------------------------------------------------
// Pass 3: out = (P0 + P1) * (1/K) + bias   (float4 per thread)
// ---------------------------------------------------------------------------
__global__ __launch_bounds__(256)
void reduce_kernel(const float* __restrict__ P,
                   const float* __restrict__ bias,
                   float* __restrict__ out) {
  const size_t i = ((size_t)blockIdx.x * 256 + threadIdx.x) * 4;
  const float4 p0 = *(const float4*)(P + i);
  const float4 p1 = *(const float4*)(P + OUT_ELEMS + i);
  const float4 b  = *(const float4*)(bias + (i & (N_DIM - 1)));
  const float inv = 1.0f / (float)K_DIM;
  float4 r;
  r.x = (p0.x + p1.x) * inv + b.x;
  r.y = (p0.y + p1.y) * inv + b.y;
  r.z = (p0.z + p1.z) * inv + b.z;
  r.w = (p0.w + p1.w) * inv + b.w;
  *(float4*)(out + i) = r;
}

// ---------------------------------------------------------------------------
// Fallback: fused convert-in-loop kernel.
// ---------------------------------------------------------------------------
__global__ __launch_bounds__(THREADS)
void feat_embed_gemm_fused_kernel(const float* __restrict__ x,
                                  const float* __restrict__ emb,
                                  const float* __restrict__ bias,
                                  float* __restrict__ out) {
  __shared__ SMem sm;

  const int t    = threadIdx.x;
  const int lane = t & 31;
  const int wid  = t >> 5;
  const int wm   = wid >> 1;
  const int wn   = wid & 1;
  const int mIn  = lane & 15;
  const int half = lane >> 4;

  const int bx = blockIdx.x;
  const int by = blockIdx.y;

  const float* gA = x   + (size_t)(by * BLK_M) * K_DIM;
  const float* gB = emb + bx * BLK_N;

  const int aIdx0 = t;
  const int aIdx1 = t + THREADS;
  const int aRow0 = aIdx0 >> 2, aCol0 = (aIdx0 & 3) * 8;
  const int aRow1 = aIdx1 >> 2, aCol1 = (aIdx1 & 3) * 8;
  const int bCol = t & 63;
  const int bK0  = (t >> 6) * 8;

  float areg[2][8];
  float breg[8];

  v8f acc[2][2];
  acc[0][0] = {}; acc[0][1] = {}; acc[1][0] = {}; acc[1][1] = {};

  auto load_global = [&](int it) {
    const int k0 = it * BLK_K;
    const float* pa0 = gA + (size_t)aRow0 * K_DIM + k0 + aCol0;
    const float* pa1 = gA + (size_t)aRow1 * K_DIM + k0 + aCol1;
    *(float4*)&areg[0][0] = ((const float4*)pa0)[0];
    *(float4*)&areg[0][4] = ((const float4*)pa0)[1];
    *(float4*)&areg[1][0] = ((const float4*)pa1)[0];
    *(float4*)&areg[1][4] = ((const float4*)pa1)[1];
#pragma unroll
    for (int j = 0; j < 8; ++j)
      breg[j] = gB[(size_t)(k0 + bK0 + j) * N_DIM + bCol];
  };

  auto stage = [&](int buf) {
#pragma unroll
    for (int c = 0; c < 2; ++c) {
      bf16x8 hv, lv;
#pragma unroll
      for (int j = 0; j < 8; ++j) {
        const float f = areg[c][j];
        const __bf16 h = (__bf16)f;
        hv[j] = h;
        lv[j] = (__bf16)(f - (float)h);
      }
      const int row = c ? aRow1 : aRow0;
      const int col = c ? aCol1 : aCol0;
      *(bf16x8*)&sm.Ahi[buf][row * LDS_K + col] = hv;
      *(bf16x8*)&sm.Alo[buf][row * LDS_K + col] = lv;
    }
    {
      bf16x8 hv, lv;
#pragma unroll
      for (int j = 0; j < 8; ++j) {
        const float f = breg[j];
        const __bf16 h = (__bf16)f;
        hv[j] = h;
        lv[j] = (__bf16)(f - (float)h);
      }
      *(bf16x8*)&sm.Bhi[buf][bCol * LDS_K + bK0] = hv;
      *(bf16x8*)&sm.Blo[buf][bCol * LDS_K + bK0] = lv;
    }
  };

  auto compute = [&](int buf) {
    const __bf16* pAhi = sm.Ahi[buf];
    const __bf16* pAlo = sm.Alo[buf];
    const __bf16* pBhi = sm.Bhi[buf];
    const __bf16* pBlo = sm.Blo[buf];

    FragBF ah[2], al[2], bh[2], bl[2];
#pragma unroll
    for (int mi = 0; mi < 2; ++mi) {
      const int r = wm * 32 + mi * 16 + mIn;
      const __bf16* ph = pAhi + r * LDS_K + half * 8;
      ah[mi].h[0] = *(const bf16x8*)(ph);
      ah[mi].h[1] = *(const bf16x8*)(ph + 16);
      const __bf16* pl = pAlo + r * LDS_K + half * 8;
      al[mi].h[0] = *(const bf16x8*)(pl);
      al[mi].h[1] = *(const bf16x8*)(pl + 16);
    }
#pragma unroll
    for (int ni = 0; ni < 2; ++ni) {
      const int n = wn * 32 + ni * 16 + mIn;
      const __bf16* ph = pBhi + n * LDS_K + half * 8;
      bh[ni].h[0] = *(const bf16x8*)(ph);
      bh[ni].h[1] = *(const bf16x8*)(ph + 16);
      const __bf16* pl = pBlo + n * LDS_K + half * 8;
      bl[ni].h[0] = *(const bf16x8*)(pl);
      bl[ni].h[1] = *(const bf16x8*)(pl + 16);
    }
#pragma unroll
    for (int mi = 0; mi < 2; ++mi) {
#pragma unroll
      for (int ni = 0; ni < 2; ++ni) {
        acc[mi][ni] = __builtin_amdgcn_wmma_f32_16x16x32_bf16(
            false, ah[mi].v, false, bh[ni].v, (short)0, acc[mi][ni], false, false);
        acc[mi][ni] = __builtin_amdgcn_wmma_f32_16x16x32_bf16(
            false, ah[mi].v, false, bl[ni].v, (short)0, acc[mi][ni], false, false);
        acc[mi][ni] = __builtin_amdgcn_wmma_f32_16x16x32_bf16(
            false, al[mi].v, false, bh[ni].v, (short)0, acc[mi][ni], false, false);
      }
    }
  };

  load_global(0);
  stage(0);
  __syncthreads();

  for (int it = 0; it < K_ITERS; ++it) {
    const int nxt = it + 1;
    if (nxt < K_ITERS) load_global(nxt);
    compute(it & 1);
    if (nxt < K_ITERS) stage(nxt & 1);
    __syncthreads();
  }

  const float inv = 1.0f / (float)K_DIM;
#pragma unroll
  for (int mi = 0; mi < 2; ++mi) {
#pragma unroll
    for (int ni = 0; ni < 2; ++ni) {
      const int row0 = by * BLK_M + wm * 32 + mi * 16 + half * 8;
      const int col  = bx * BLK_N + wn * 32 + ni * 16 + mIn;
      const float bv = bias[col];
#pragma unroll
      for (int j = 0; j < 8; ++j)
        out[(size_t)(row0 + j) * N_DIM + col] = acc[mi][ni][j] * inv + bv;
    }
  }
}

// ---------------------------------------------------------------------------
extern "C" void kernel_launch(void* const* d_in, const int* in_sizes, int n_in,
                              void* d_out, int out_size, void* d_ws, size_t ws_size,
                              hipStream_t stream) {
  (void)in_sizes; (void)n_in; (void)out_size;
  const float* x    = (const float*)d_in[0];
  const float* emb  = (const float*)d_in[1];
  const float* bias = (const float*)d_in[2];
  float* out        = (float*)d_out;

  if (ws_size >= WS_NEED) {
    __bf16* Ahi = (__bf16*)d_ws;
    __bf16* Alo = Ahi + A_ELEMS;
    __bf16* Bhi = Alo + A_ELEMS;
    __bf16* Blo = Bhi + B_ELEMS;
    float*  P   = (float*)((char*)d_ws + WS_BF16);

    split_x_kernel<<<(int)(A_ELEMS / (256 * 8)), 256, 0, stream>>>(x, Ahi, Alo);
    split_et_kernel<<<dim3(K_DIM / 32, N_DIM / 64), 256, 0, stream>>>(emb, Bhi, Blo);
    gemm_bf16_async_kernel<<<dim3(N_DIM / BLK_N, M_DIM / BLK_M, SPLIT_K),
                             THREADS, 0, stream>>>(Ahi, Alo, Bhi, Blo, P);
    reduce_kernel<<<(int)(OUT_ELEMS / (256 * 4)), 256, 0, stream>>>(P, bias, out);
  } else {
    const dim3 grid(N_DIM / BLK_N, M_DIM / BLK_M);  // (8, 16)
    feat_embed_gemm_fused_kernel<<<grid, THREADS, 0, stream>>>(x, emb, bias, out);
  }
}
